// FBRNN_86904368267680
// MI455X (gfx1250) — matrix operand order repaired and checked by reading, not compiled, not run on hardware
//
#include <hip/hip_runtime.h>
#include <hip/hip_bf16.h>

#define H 16
#define LL 4
#define F 256
#define TT 131072

typedef __attribute__((ext_vector_type(16))) _Float16 v16h;
typedef __attribute__((ext_vector_type(8)))  float    v8f;

// ---------------------------------------------------------------------------
// Kernel A: G0[t][g*16+n] = batch[t,:] @ Wih0[g*16+n,:] + bih0   (t in [0,T))
// One wave per 16-row t-tile, 3 N-tiles (48 outputs), K looped 256 in 32s.
// Split-f16 (Markidis) compensation: 3 WMMAs per tile-k to recover ~f32 acc.
// ---------------------------------------------------------------------------
__global__ __launch_bounds__(256) void g0_gemm_kernel(
    const float* __restrict__ X,      // [T,256]
    const float* __restrict__ W,      // [48,256]  (Wih0)
    const float* __restrict__ bias,   // [48]
    float* __restrict__ G0)           // [T,48]
{
  const int lane   = threadIdx.x & 31;
  const int wave   = threadIdx.x >> 5;
  const int tile   = blockIdx.x * 8 + wave;     // 8192 tiles total
  const int t0     = tile * 16;
  const int mrow   = lane & 15;                 // A-matrix row / C column
  const int hiHalf = (lane >> 4) & 1;

  v8f acc0 = {}; v8f acc1 = {}; v8f acc2 = {};

  const float* xrow = X + (size_t)(t0 + mrow) * F;

  for (int k0 = 0; k0 < F; k0 += 32) {
    // ---- A fragment (16x32 f16): lane holds row mrow; half j -> K = j + 16*hi
    v16h a_hi, a_lo;
#pragma unroll
    for (int j = 0; j < 16; ++j) {
      int K = j + hiHalf * 16;
      float x = xrow[k0 + K];
      _Float16 h = (_Float16)x;
      a_hi[j] = h;
      a_lo[j] = (_Float16)(x - (float)h);
    }
    // ---- B fragments (32x16 f16): lane holds col n; half j -> K = j + 16*hi
#pragma unroll
    for (int nt = 0; nt < 3; ++nt) {
      const int n = nt * 16 + mrow;             // output column within 48
      const float* wrow = W + (size_t)n * F + k0 + hiHalf * 16;
      v16h b_hi, b_lo;
#pragma unroll
      for (int j = 0; j < 16; ++j) {
        float w = wrow[j];
        _Float16 h = (_Float16)w;
        b_hi[j] = h;
        b_lo[j] = (_Float16)(w - (float)h);
      }
      v8f c = (nt == 0) ? acc0 : (nt == 1) ? acc1 : acc2;
      c = __builtin_amdgcn_wmma_f32_16x16x32_f16(false, a_hi, false, b_hi,
                                                 (short)0, c, false, false);
      c = __builtin_amdgcn_wmma_f32_16x16x32_f16(false, a_hi, false, b_lo,
                                                 (short)0, c, false, false);
      c = __builtin_amdgcn_wmma_f32_16x16x32_f16(false, a_lo, false, b_hi,
                                                 (short)0, c, false, false);
      if (nt == 0) acc0 = c; else if (nt == 1) acc1 = c; else acc2 = c;
    }
  }

  // ---- store C/D: VGPR r -> M = r + 8*hiHalf ; N = lane&15 ; add bias
#pragma unroll
  for (int nt = 0; nt < 3; ++nt) {
    const int n = nt * 16 + mrow;
    const float bn = bias[n];
    const v8f c = (nt == 0) ? acc0 : (nt == 1) ? acc1 : acc2;
#pragma unroll
    for (int r = 0; r < 8; ++r) {
      int t = t0 + r + 8 * hiHalf;
      G0[(size_t)t * 48 + n] = c[r] + bn;
    }
  }
}

// ---------------------------------------------------------------------------
// Kernel B: the sequential scan. One wave32 workgroup; weights+state in LDS.
// Lane j (<16) owns hidden unit j of the active layer. G0 row prefetched.
// ---------------------------------------------------------------------------
__device__ __forceinline__ float wave_sum32(float v) {
  v += __shfl_xor(v, 1, 32);
  v += __shfl_xor(v, 2, 32);
  v += __shfl_xor(v, 4, 32);
  v += __shfl_xor(v, 8, 32);
  v += __shfl_xor(v, 16, 32);
  return v;
}
__device__ __forceinline__ float sigmoidf_(float x) {
  return 1.0f / (1.0f + __expf(-x));
}

__global__ __launch_bounds__(32) void rnn_scan_kernel(
    const float* __restrict__ G0,     // [T,48] precomputed layer0 gi
    const float* __restrict__ Whh0,   // [48,16]
    const float* __restrict__ bhh0,   // [48]
    const float* __restrict__ WihL,   // [3,48,16]
    const float* __restrict__ WhhL,   // [3,48,16]
    const float* __restrict__ bihL,   // [3,48]
    const float* __restrict__ bhhL,   // [3,48]
    const float* __restrict__ aW,     // [4,16,16]
    const float* __restrict__ ab,     // [4,16]
    const float* __restrict__ av,     // [4,16]
    const float* __restrict__ avb,    // [4]
    const float* __restrict__ fc1W,   // [32,16]
    const float* __restrict__ fc1b,   // [32]
    const float* __restrict__ fc2W,   // [32]
    const float* __restrict__ fc2b,   // [1]
    float* __restrict__ out)          // [T]
{
  __shared__ float sWhh0[768];
  __shared__ float sbhh0[48];
  __shared__ float sWih[2304];
  __shared__ float sWhh[2304];
  __shared__ float sbih[144];
  __shared__ float sbhh[144];
  __shared__ float saW[1024];
  __shared__ float sab[64];
  __shared__ float sav[64];
  __shared__ float savb[4];
  __shared__ float sfc1W[512];
  __shared__ float sfc1b[32];
  __shared__ float sfc2W[32];
  __shared__ float hcur[64];   // hs[l*16+j]
  __shared__ float stk[64];    // per-step GRU outputs (stack)
  __shared__ float hsn[64];    // per-step attention outputs (hs_new)

  const int j   = threadIdx.x;   // 0..31
  const int j16 = j & 15;
  const bool act = (j < 16);

  for (int i = j; i <  768; i += 32) sWhh0[i] = Whh0[i];
  for (int i = j; i <   48; i += 32) sbhh0[i] = bhh0[i];
  for (int i = j; i < 2304; i += 32) sWih[i]  = WihL[i];
  for (int i = j; i < 2304; i += 32) sWhh[i]  = WhhL[i];
  for (int i = j; i <  144; i += 32) sbih[i]  = bihL[i];
  for (int i = j; i <  144; i += 32) sbhh[i]  = bhhL[i];
  for (int i = j; i < 1024; i += 32) saW[i]   = aW[i];
  for (int i = j; i <   64; i += 32) sab[i]   = ab[i];
  for (int i = j; i <   64; i += 32) sav[i]   = av[i];
  for (int i = j; i <    4; i += 32) savb[i]  = avb[i];
  for (int i = j; i <  512; i += 32) sfc1W[i] = fc1W[i];
  for (int i = j; i <   32; i += 32) sfc1b[i] = fc1b[i];
  for (int i = j; i <   32; i += 32) sfc2W[i] = fc2W[i];
  for (int i = j; i <   64; i += 32) { hcur[i] = 0.0f; stk[i] = 0.0f; hsn[i] = 0.0f; }
  const float c_fc2b = fc2b[0];
  __syncthreads();   // 1-wave workgroup: lowers to S_NOP (free)

  // prefetch G0 row for t=0
  float ngi0 = G0[j16];
  float ngi1 = G0[16 + j16];
  float ngi2 = G0[32 + j16];

  for (int t = 0; t < TT; ++t) {
    const float gi0 = ngi0, gi1 = ngi1, gi2 = ngi2;
    const int tn = (t + 1 < TT) ? t + 1 : t;
    ngi0 = G0[(size_t)tn * 48 + j16];          // issued early; latency hidden
    ngi1 = G0[(size_t)tn * 48 + 16 + j16];
    ngi2 = G0[(size_t)tn * 48 + 32 + j16];

    // ---------------- layer 0 ----------------
    {
      float ghr = sbhh0[j16], ghz = sbhh0[16 + j16], ghn = sbhh0[32 + j16];
#pragma unroll
      for (int m = 0; m < 16; ++m) {
        const float hm = hcur[m];
        ghr = fmaf(hm, sWhh0[j16 * 16 + m], ghr);
        ghz = fmaf(hm, sWhh0[(16 + j16) * 16 + m], ghz);
        ghn = fmaf(hm, sWhh0[(32 + j16) * 16 + m], ghn);
      }
      const float r = sigmoidf_(gi0 + ghr);
      const float z = sigmoidf_(gi1 + ghz);
      const float n = tanhf(gi2 + r * ghn);
      const float hn = (1.0f - z) * n + z * hcur[j16];
      if (act) stk[j16] = hn;
    }
    __syncthreads();

    // ---------------- layers 1..3 ----------------
#pragma unroll
    for (int l = 1; l < LL; ++l) {
      const float* wi = sWih + (l - 1) * 768;
      const float* wh = sWhh + (l - 1) * 768;
      const float* bi = sbih + (l - 1) * 48;
      const float* bh = sbhh + (l - 1) * 48;
      float gir = bi[j16], giz = bi[16 + j16], gin = bi[32 + j16];
      float ghr = bh[j16], ghz = bh[16 + j16], ghn = bh[32 + j16];
#pragma unroll
      for (int m = 0; m < 16; ++m) {
        const float xm = stk[(l - 1) * 16 + m];
        const float hm = hcur[l * 16 + m];
        gir = fmaf(xm, wi[j16 * 16 + m], gir);
        giz = fmaf(xm, wi[(16 + j16) * 16 + m], giz);
        gin = fmaf(xm, wi[(32 + j16) * 16 + m], gin);
        ghr = fmaf(hm, wh[j16 * 16 + m], ghr);
        ghz = fmaf(hm, wh[(16 + j16) * 16 + m], ghz);
        ghn = fmaf(hm, wh[(32 + j16) * 16 + m], ghn);
      }
      const float r = sigmoidf_(gir + ghr);
      const float z = sigmoidf_(giz + ghz);
      const float n = tanhf(gin + r * ghn);
      const float hn = (1.0f - z) * n + z * hcur[l * 16 + j16];
      if (act) stk[l * 16 + j16] = hn;
      __syncthreads();
    }

    // ---------------- attention over layers ----------------
#pragma unroll
    for (int i = 0; i < LL; ++i) {
      float s[LL];
      for (int k = i; k < LL; ++k) {
        float acc = sab[i * 16 + j16];
#pragma unroll
        for (int m = 0; m < 16; ++m)
          acc = fmaf(stk[k * 16 + m], saW[i * 256 + m * 16 + j16], acc);
        float tk = tanhf(acc) * sav[i * 16 + j16];
        tk = act ? tk : 0.0f;
        s[k] = wave_sum32(tk) + savb[i];
      }
      float mx = s[i];
      for (int k = i + 1; k < LL; ++k) mx = fmaxf(mx, s[k]);
      float e[LL]; float den = 0.0f;
      for (int k = i; k < LL; ++k) { e[k] = __expf(s[k] - mx); den += e[k]; }
      const float inv = 1.0f / den;
      float o = 0.0f;
      for (int k = i; k < LL; ++k) o = fmaf(e[k] * inv, stk[k * 16 + j16], o);
      if (act) hsn[i * 16 + j16] = o;
    }
    __syncthreads();

    // hs = hs_new for next step
    hcur[j] = hsn[j];
    hcur[j + 32] = hsn[j + 32];
    __syncthreads();

    // ---------------- FC head on top layer output ----------------
    float a1 = sfc1b[j];
#pragma unroll
    for (int m = 0; m < 16; ++m)
      a1 = fmaf(hsn[48 + m], sfc1W[j * 16 + m], a1);
    a1 = fmaxf(a1, 0.0f);
    float v = wave_sum32(a1 * sfc2W[j]);
    if (j == 0) out[t] = v + c_fc2b;
    __syncthreads();
  }
}

// ---------------------------------------------------------------------------
extern "C" void kernel_launch(void* const* d_in, const int* in_sizes, int n_in,
                              void* d_out, int out_size, void* d_ws, size_t ws_size,
                              hipStream_t stream) {
  const float* batch = (const float*)d_in[0];
  const float* Wih0  = (const float*)d_in[1];
  const float* Whh0  = (const float*)d_in[2];
  const float* bih0  = (const float*)d_in[3];
  const float* bhh0  = (const float*)d_in[4];
  const float* WihL  = (const float*)d_in[5];
  const float* WhhL  = (const float*)d_in[6];
  const float* bihL  = (const float*)d_in[7];
  const float* bhhL  = (const float*)d_in[8];
  const float* aW    = (const float*)d_in[9];
  const float* ab    = (const float*)d_in[10];
  const float* av    = (const float*)d_in[11];
  const float* avb   = (const float*)d_in[12];
  const float* fc1W  = (const float*)d_in[13];
  const float* fc1b  = (const float*)d_in[14];
  const float* fc2W  = (const float*)d_in[15];
  const float* fc2b  = (const float*)d_in[16];
  float* G0 = (float*)d_ws;              // T*48 floats = ~25.2 MB scratch
  float* out = (float*)d_out;

  // Kernel A: 8192 wave-tiles, 8 waves (256 thr) per block -> 1024 blocks
  g0_gemm_kernel<<<dim3(TT / (16 * 8)), dim3(256), 0, stream>>>(batch, Wih0, bih0, G0);

  // Kernel B: one wave runs the whole sequential scan
  rnn_scan_kernel<<<dim3(1), dim3(32), 0, stream>>>(G0, Whh0, bhh0, WihL, WhhL,
                                                    bihL, bhhL, aW, ab, av, avb,
                                                    fc1W, fc1b, fc2W, fc2b, out);
}
